// SwinTransformerBlock_20203526160520
// MI455X (gfx1250) — matrix operand order repaired and checked by reading, not compiled, not run on hardware
//
#include <hip/hip_runtime.h>
#include <hip/hip_bf16.h>

typedef __bf16 bf16;
typedef __attribute__((ext_vector_type(16))) __bf16 v16bf;
typedef __attribute__((ext_vector_type(8)))  float  v8f;

#define HH_   56
#define WW_   56
#define WS_   7
#define SS_   3
#define HEADS 6
#define DIM   192
#define HID   768
#define BATCH 32
#define NTOK  49      // tokens per window
#define NWIN  64      // windows per image
#define ROWS  (BATCH*HH_*WW_)   // 100352 token rows

// ---------------------------------------------------------------- utilities

__global__ void cvt_bf16_kernel(const float* __restrict__ in, bf16* __restrict__ out, int n) {
  int i = blockIdx.x * 256 + threadIdx.x;
  if (i < n) out[i] = (bf16)in[i];
}

// LayerNorm over DIM=192, one row per block (192 threads, 6 waves).
// mode 1: gather source row through (shift + window-partition) mapping (LN1 path).
// mode 0: identity rows (LN2 path).
__global__ __launch_bounds__(192) void ln_kernel(
    const float* __restrict__ x, const float* __restrict__ g, const float* __restrict__ b,
    bf16* __restrict__ out, int mode) {
  int r = blockIdx.x;
  int t = threadIdx.x;
  int src;
  if (mode == 1) {
    int bb  = r / (NWIN * NTOK);
    int rem = r % (NWIN * NTOK);
    int w = rem / NTOK, n = rem % NTOK;
    int hh = (w >> 3) * WS_ + n / WS_;
    int ww = (w & 7) * WS_ + n % WS_;
    int sh = (hh + SS_) % HH_, sw = (ww + SS_) % WW_;
    src = bb * (HH_ * WW_) + sh * WW_ + sw;
  } else {
    src = r;
  }
  float v = x[(long)src * DIM + t];
  __shared__ float s[256], s2[256];
  s[t] = v; s2[t] = v * v;
  if (t < 64) { s[192 + t] = 0.f; s2[192 + t] = 0.f; }
  __syncthreads();
  for (int off = 128; off > 0; off >>= 1) {
    if (t < off) { s[t] += s[t + off]; s2[t] += s2[t + off]; }
    __syncthreads();
  }
  float mu  = s[0] * (1.f / DIM);
  float var = s2[0] * (1.f / DIM) - mu * mu;
  float inv = rsqrtf(var + 1e-5f);
  out[(long)r * DIM + t] = (bf16)((v - mu) * inv * g[t] + b[t]);
}

// -------------------------------------------------- generic WMMA GEMM
// C[M,N] = A[M,K](bf16) * B[K,N](bf16) + bias.
// Each block owns 64 rows and iterates over ALL N-tiles (NT = N/64), so the
// A panel is read from HBM exactly once per K-chunk (A-traffic dominates:
// weights are L2-resident on a 192MB-L2 part). A panel (64 x 192, stride 200,
// conflict-free) staged in LDS per chunk; its 6 fragments hoisted to registers
// and reused by every N-tile. Accumulators persist in VGPRs across K-chunks.
// EPI 0: out bf16              EPI 1: out bf16 after exact GELU
// EPI 2: out f32 = v + Res[row*N+col]
// EPI 3: proj: window-reverse + unshift scatter, out f32 = v + Res[dest]
template <int EPI, int NT>
__global__ __launch_bounds__(256) void gemm_wmma(
    const bf16* __restrict__ A, const bf16* __restrict__ Bw,
    const float* __restrict__ bias, void* __restrict__ Out,
    const float* __restrict__ Res, int M, int N, int K) {
  __shared__ bf16 As[64 * 200];   // 64 rows x 192 K, stride 200
  __shared__ bf16 Bs[192 * 64];   // 192 K x 64 cols
  int t = threadIdx.x, lane = t & 31, wid = t >> 5;
  int wm = wid & 3, wn = wid >> 2;          // 4 M-subtiles x 2 N-halves (16x32 per wave)
  int lrow = lane & 15, hf = lane >> 4;
  long rowBase = (long)blockIdx.x * 64;

  v8f zero = {0.f, 0.f, 0.f, 0.f, 0.f, 0.f, 0.f, 0.f};
  v8f acc[NT][2];
#pragma unroll
  for (int i = 0; i < NT; ++i) { acc[i][0] = zero; acc[i][1] = zero; }

  for (int kc = 0; kc < K; kc += 192) {
    // ---- stage A panel (64x192): 6 b128 loads/stores per thread
#pragma unroll
    for (int it = 0; it < 6; ++it) {
      int idx = it * 256 + t;
      int r = idx / 24, c = (idx % 24) * 8;
      *(uint4*)(As + r * 200 + c) =
          *(const uint4*)(A + (rowBase + r) * K + kc + c);
    }
    __syncthreads();
    // ---- hoist all 6 A fragments for this chunk into registers
    v16bf af[6];
#pragma unroll
    for (int ki = 0; ki < 6; ++ki) {
      const bf16* ap = As + (wm * 16 + lrow) * 200 + ki * 32;
#pragma unroll
      for (int j = 0; j < 16; ++j)
        af[ki][j] = ap[hf * 16 + j];
    }
#pragma unroll
    for (int nt = 0; nt < NT; ++nt) {
      // ---- stage B panel (192x64) for this N-tile
#pragma unroll
      for (int it = 0; it < 6; ++it) {
        int idx = it * 256 + t;
        int r = idx >> 3, c = (idx & 7) * 8;
        *(uint4*)(Bs + r * 64 + c) =
            *(const uint4*)(Bw + (long)(kc + r) * N + nt * 64 + c);
      }
      __syncthreads();
#pragma unroll
      for (int ki = 0; ki < 6; ++ki) {
        v16bf b0, b1;
        const bf16* bp = Bs + (ki * 32 + hf * 16) * 64 + wn * 32 + lrow;
#pragma unroll
        for (int j = 0; j < 16; ++j) { b0[j] = bp[j * 64]; b1[j] = bp[j * 64 + 16]; }
        acc[nt][0] = __builtin_amdgcn_wmma_f32_16x16x32_bf16(
            false, af[ki], false, b0, (short)0, acc[nt][0], false, false);
        acc[nt][1] = __builtin_amdgcn_wmma_f32_16x16x32_bf16(
            false, af[ki], false, b1, (short)0, acc[nt][1], false, false);
      }
      __syncthreads();   // Bs (and As at chunk end) safe to restage
    }
  }

  // ---- epilogue: D layout -> VGPR r holds M = r + 8*(lane/16), N = lane%16
  int rbase = wm * 16 + (hf << 3);
#pragma unroll
  for (int nt = 0; nt < NT; ++nt) {
    int cb0 = nt * 64 + wn * 32 + lrow;
#pragma unroll
    for (int r = 0; r < 8; ++r) {
      long row = rowBase + rbase + r;
#pragma unroll
      for (int s = 0; s < 2; ++s) {
        int col = cb0 + s * 16;
        float v = acc[nt][s][r] + bias[col];
        if (EPI == 0) {
          ((bf16*)Out)[row * N + col] = (bf16)v;
        } else if (EPI == 1) {
          v = 0.5f * v * (1.f + erff(v * 0.70710678118654752f));
          ((bf16*)Out)[row * N + col] = (bf16)v;
        } else if (EPI == 2) {
          ((float*)Out)[row * N + col] = v + Res[row * N + col];
        } else {
          int bb  = (int)(row / (NWIN * NTOK));
          int rem = (int)(row % (NWIN * NTOK));
          int w = rem / NTOK, n = rem % NTOK;
          int hh = (w >> 3) * WS_ + n / WS_;
          int ww = (w & 7) * WS_ + n % WS_;
          int dh = (hh + SS_) % HH_, dw = (ww + SS_) % WW_;
          long dr = (long)bb * (HH_ * WW_) + dh * WW_ + dw;
          ((float*)Out)[dr * DIM + col] = v + Res[dr * DIM + col];
        }
      }
    }
  }
}

// -------------------------------------------------- windowed attention
// One wave per (window-batch b_, head). qkv is bf16 [2048*49, 576],
// cols: q = head*32+k, k-mat = 192+head*32+k, v = 384+head*32+k.
//
// Per-wave LDS layout (14848 B):
//   [0 .. 4608)      : Q tile (64 x 32 bf16, stride 36)  -- later reused for V tile
//   [4608 .. 14848)  : K tile (64 x 32 bf16, stride 36)  -- later reused for scores S
//                      S: f32 stride 52 (52*r mod 64 distinct -> conflict-free),
//                      then probabilities bf16 in place (stride 104), pads zeroed.
#define T_STRIDE 36
#define S_STRIDE 52
#define S_ELEMS  (48 * S_STRIDE + 64)             // padded col tiles stay in-bounds
#define WAVE_LDS (64 * T_STRIDE * 2 + S_ELEMS * 4) // 4608 + 10240 = 14848

// Stage a 49x32 bf16 tile (row stride 576 in global) into LDS rows 0..63
// (rows 49..63 zero-filled). Pure b128 global loads, b64 LDS stores.
__device__ inline void load_tile49(const bf16* __restrict__ g, bf16* lds, int lane) {
#pragma unroll
  for (int it = 0; it < 4; ++it) {
    int idx = it * 32 + lane;
    int row = idx >> 1, c = (idx & 1) << 4;       // 2 lanes per row, 16 cols each
    uint4 a = {0, 0, 0, 0}, b = {0, 0, 0, 0};
    if (row < NTOK) {
      const uint4* gp = (const uint4*)(g + (long)row * 576 + c);
      a = gp[0];
      b = gp[1];
    }
    uint2* lp = (uint2*)(lds + row * T_STRIDE + c);
    lp[0] = make_uint2(a.x, a.y);
    lp[1] = make_uint2(a.z, a.w);
    lp[2] = make_uint2(b.x, b.y);
    lp[3] = make_uint2(b.z, b.w);
  }
}

__global__ __launch_bounds__(128) void attn_kernel(
    const bf16* __restrict__ qkv, const float* __restrict__ rpb,
    bf16* __restrict__ out) {
  __shared__ __align__(16) char smem[4][WAVE_LDS];
  int lane = threadIdx.x & 31, wid = threadIdx.x >> 5;
  int task = blockIdx.x * 4 + wid;
  int b_ = task / HEADS, head = task % HEADS;
  int w  = b_ & 63;
  bf16*  TQ = (bf16*)smem[wid];                      // Q tile, later V tile
  bf16*  TK = (bf16*)(smem[wid] + 64 * T_STRIDE * 2);// K tile
  float* S  = (float*)(smem[wid] + 64 * T_STRIDE * 2); // scores (overlap TK)
  bf16*  P  = (bf16*)S;                              // probs bf16 in place, stride 104
  int lrow = lane & 15, hf = lane >> 4;

  const bf16* base = qkv + (long)b_ * NTOK * 576 + head * 32;

  // phase 1: stage Q and K tiles
  load_tile49(base, TQ, lane);
  load_tile49(base + 192, TK, lane);
  __syncthreads();

  // phase 2: hoist all Q (A) and K (B) fragments to registers (no guards)
  v16bf qa[4], kb[4];
#pragma unroll
  for (int mt = 0; mt < 4; ++mt) {
    const bf16* p = TQ + (mt * 16 + lrow) * T_STRIDE;
#pragma unroll
    for (int j = 0; j < 16; ++j)
      qa[mt][j] = p[hf * 16 + j];
  }
#pragma unroll
  for (int nt = 0; nt < 4; ++nt) {
    const bf16* p = TK + (nt * 16 + lrow) * T_STRIDE + hf * 16;
#pragma unroll
    for (int j = 0; j < 16; ++j)
      kb[nt][j] = p[j];
  }
  __syncthreads();

  // phase 3: stage V tile into TQ region; compute Q@K^T tiles into S (over TK)
  load_tile49(base + 384, TQ, lane);
#pragma unroll
  for (int mt = 0; mt < 4; ++mt) {
#pragma unroll
    for (int nt = 0; nt < 4; ++nt) {
      v8f c = {};
      c = __builtin_amdgcn_wmma_f32_16x16x32_bf16(false, qa[mt], false, kb[nt],
                                                  (short)0, c, false, false);
#pragma unroll
      for (int r = 0; r < 8; ++r) {
        int rr = mt * 16 + hf * 8 + r;
        if (rr < NTOK) S[rr * S_STRIDE + nt * 16 + lrow] = c[r];
      }
    }
  }
  __syncthreads();

  // phase 4: scale + rel-pos bias + shift mask + softmax; convert to bf16 in place
  const float scale = 0.17677669529663687f;   // 1/sqrt(32)
  int wh = w >> 3, wwj = w & 7;
  for (int n = lane; n < NTOK; n += 32) {
    int ny = n / WS_, nx = n % WS_;
    int hn = wh * WS_ + ny, wn_ = wwj * WS_ + nx;
    int gn = (hn < 49 ? 0 : (hn < 53 ? 1 : 2)) * 3 + (wn_ < 49 ? 0 : (wn_ < 53 ? 1 : 2));
    float* rowp = S + n * S_STRIDE;
    float mx = -1e30f;
    for (int m = 0; m < NTOK; ++m) {
      int my = m / WS_, mxc = m % WS_;
      int hm = wh * WS_ + my, wm_ = wwj * WS_ + mxc;
      int gm = (hm < 49 ? 0 : (hm < 53 ? 1 : 2)) * 3 + (wm_ < 49 ? 0 : (wm_ < 53 ? 1 : 2));
      int dy = ny - my + 6, dx = nx - mxc + 6;
      float v = rowp[m] * scale + rpb[(dy * 13 + dx) * HEADS + head]
              + ((gn != gm) ? -100.f : 0.f);
      rowp[m] = v;
      mx = fmaxf(mx, v);
    }
    float sum = 0.f;
    for (int m = 0; m < NTOK; ++m) { float e = expf(rowp[m] - mx); rowp[m] = e; sum += e; }
    float inv = 1.f / sum;
    bf16* pb = P + n * (S_STRIDE * 2);
    for (int m = 0; m < NTOK; ++m) pb[m] = (bf16)(rowp[m] * inv);  // write lags read: safe
#pragma unroll
    for (int m = NTOK; m < 64; ++m) pb[m] = (bf16)0.f;             // zero K-pad slots
  }
  __syncthreads();

  // phase 5: P @ V (K = 49 padded to 64, 2 K-steps), V from LDS, P from LDS
  for (int mt = 0; mt < 4; ++mt) {
    int row = mt * 16 + lrow;
    for (int nt = 0; nt < 2; ++nt) {
      v8f c = {};
#pragma unroll
      for (int ks = 0; ks < 2; ++ks) {
        v16bf a;
        if (row < NTOK) {
          const bf16* pr = P + row * (S_STRIDE * 2) + ks * 32;
#pragma unroll
          for (int j = 0; j < 16; ++j)
            a[j] = pr[hf * 16 + j];
        } else {
#pragma unroll
          for (int j = 0; j < 16; ++j) a[j] = (bf16)0.f;
        }
        v16bf vb;
        const bf16* vp = TQ + (ks * 32 + hf * 16) * T_STRIDE + nt * 16 + lrow;
#pragma unroll
        for (int j = 0; j < 16; ++j)
          vb[j] = vp[j * T_STRIDE];
        c = __builtin_amdgcn_wmma_f32_16x16x32_bf16(false, a, false, vb,
                                                    (short)0, c, false, false);
      }
#pragma unroll
      for (int r = 0; r < 8; ++r) {
        int rr = mt * 16 + hf * 8 + r;
        if (rr < NTOK)
          out[((long)b_ * NTOK + rr) * DIM + head * 32 + nt * 16 + lrow] = (bf16)c[r];
      }
    }
  }
}

// -------------------------------------------------- launch
extern "C" void kernel_launch(void* const* d_in, const int* in_sizes, int n_in,
                              void* d_out, int out_size, void* d_ws, size_t ws_size,
                              hipStream_t stream) {
  const float* x       = (const float*)d_in[0];
  const float* n1g     = (const float*)d_in[1];
  const float* n1b     = (const float*)d_in[2];
  const float* qkv_w   = (const float*)d_in[3];
  const float* qkv_b   = (const float*)d_in[4];
  const float* rpb     = (const float*)d_in[5];
  const float* proj_w  = (const float*)d_in[6];
  const float* proj_b  = (const float*)d_in[7];
  const float* n2g     = (const float*)d_in[8];
  const float* n2b     = (const float*)d_in[9];
  const float* fc1_w   = (const float*)d_in[10];
  const float* fc1_b   = (const float*)d_in[11];
  const float* fc2_w   = (const float*)d_in[12];
  const float* fc2_b   = (const float*)d_in[13];

  size_t off = 0;
  auto carve = [&](size_t bytes) {
    void* p = (char*)d_ws + off;
    off += (bytes + 255) & ~(size_t)255;
    return p;
  };
  bf16*  xw   = (bf16*)carve((size_t)ROWS * DIM * 2);          // LN1 out, reused for LN2 out
  bf16*  big  = (bf16*)carve((size_t)ROWS * HID * 2);          // qkv (576), reused for MLP hidden (768)
  bf16*  aout = (bf16*)carve((size_t)ROWS * DIM * 2);          // attention output (pre-proj)
  float* x1   = (float*)carve((size_t)ROWS * DIM * 4);         // residual 1 (spatial order)
  bf16*  qw_b = (bf16*)carve((size_t)DIM * 3 * DIM * 2);
  bf16*  pw_b = (bf16*)carve((size_t)DIM * DIM * 2);
  bf16*  w1_b = (bf16*)carve((size_t)DIM * HID * 2);
  bf16*  w2_b = (bf16*)carve((size_t)HID * DIM * 2);

  // weight conversion f32 -> bf16
  cvt_bf16_kernel<<<(DIM * 3 * DIM + 255) / 256, 256, 0, stream>>>(qkv_w, qw_b, DIM * 3 * DIM);
  cvt_bf16_kernel<<<(DIM * DIM + 255) / 256, 256, 0, stream>>>(proj_w, pw_b, DIM * DIM);
  cvt_bf16_kernel<<<(DIM * HID + 255) / 256, 256, 0, stream>>>(fc1_w, w1_b, DIM * HID);
  cvt_bf16_kernel<<<(HID * DIM + 255) / 256, 256, 0, stream>>>(fc2_w, w2_b, HID * DIM);

  // LN1 + shift + window partition
  ln_kernel<<<ROWS, 192, 0, stream>>>(x, n1g, n1b, xw, 1);
  // QKV projection: N = 576 -> 9 N-tiles
  gemm_wmma<0, 9><<<ROWS / 64, 256, 0, stream>>>(
      xw, qw_b, qkv_b, big, nullptr, ROWS, 576, DIM);
  // windowed attention (2048 window-batches x 6 heads, 4 waves/block)
  attn_kernel<<<(BATCH * NWIN * HEADS) / 4, 128, 0, stream>>>(big, rpb, aout);
  // output projection + window reverse + unshift + residual: N = 192 -> 3 N-tiles
  gemm_wmma<3, 3><<<ROWS / 64, 256, 0, stream>>>(
      aout, pw_b, proj_b, x1, x, ROWS, DIM, DIM);
  // LN2
  ln_kernel<<<ROWS, 192, 0, stream>>>(x1, n2g, n2b, xw, 0);
  // MLP: fc1 N = 768 -> 12 N-tiles; fc2 K = 768 (4 K-chunks), N = 192 -> 3 N-tiles
  gemm_wmma<1, 12><<<ROWS / 64, 256, 0, stream>>>(
      xw, w1_b, fc1_b, big, nullptr, ROWS, HID, DIM);
  gemm_wmma<2, 3><<<ROWS / 64, 256, 0, stream>>>(
      big, w2_b, fc2_b, d_out, x1, ROWS, DIM, HID);
}
